// Baseline_GNN_10256381903665
// MI455X (gfx1250) — matrix operand order repaired and checked by reading, not compiled, not run on hardware
//
#include <hip/hip_runtime.h>
#include <math.h>

typedef __attribute__((ext_vector_type(16))) _Float16 v16h;
typedef __attribute__((ext_vector_type(8)))  _Float16 v8h;
typedef __attribute__((ext_vector_type(4)))  _Float16 v4h;
typedef __attribute__((ext_vector_type(8)))  float    v8f;

#define T_DIM   512
#define ROI     200
#define BATCH   256
#define MROWS   (BATCH * ROI)      // 51200
#define ROI_PAD 224
#define NLAYERS 3
#define BN_EPS  1e-5f

#define LDSA 40   // halves per row in LDS A tile (16B aligned, bank-rotating)
#define LDSB 40   // halves per row in LDS B^T tile
#define LDST 520  // halves per row in transpose-staging tile (bank-rotating)

// ---------------- workspace layout (bytes, all 256-aligned) ----------------
#define OFF_H   ((size_t)0)           // 51200x512 f32            (104857600)
#define OFF_U   ((size_t)104857600)   // 51200x512 f32; HPT aliases here
#define OFF_V16 ((size_t)209715200)   // 51200x512 f16; T16 aliases here
#define OFF_MP  ((size_t)262144000)   // 256x224x224 f16 padded mask
#define OFF_W16 ((size_t)287834112)   // 512x512 f16 transposed weight staging
#define OFF_PS  ((size_t)288358400)   // 400x512 f32 partial sums
#define OFF_PQ  ((size_t)289177600)   // 400x512 f32 partial sumsq
#define OFF_SS  ((size_t)289996800)   // 2x512 f32 scale/shift
#define OFF_XM  ((size_t)290000896)   // 256x512 f32
#define OFF_Y1  ((size_t)290525184)   // 256x256 f32
#define OFF_Z   ((size_t)290787328)   // 256x256 f32

// ---------------- WMMA fragment builders (wave32, 16-bit layouts) ----------
// A (16x32, row-major in LDS): lane<16 -> M=lane, k0=0; lane>=16 -> M=lane-16, k0=8
// halves {k0..k0+7, 16+k0..16+k0+7}
__device__ __forceinline__ v16h load_a_frag(const _Float16* sA, int row0) {
    int lane = threadIdx.x & 31;
    int m    = lane & 15;
    int k0   = (lane & 16) ? 8 : 0;
    const _Float16* p = sA + (row0 + m) * LDSA + k0;
    union { v16h v; v8h h[2]; } u;
    u.h[0] = *reinterpret_cast<const v8h*>(p);
    u.h[1] = *reinterpret_cast<const v8h*>(p + 16);
    return u.v;
}
// B (32x16) from N-major LDS tile: lane = column, K contiguous.
__device__ __forceinline__ v16h load_b_frag(const _Float16* sBT, int col0) {
    int lane = threadIdx.x & 31;
    int n    = lane & 15;
    int k0   = (lane & 16) ? 8 : 0;
    const _Float16* p = sBT + (col0 + n) * LDSB + k0;
    union { v16h v; v8h h[2]; } u;
    u.h[0] = *reinterpret_cast<const v8h*>(p);
    u.h[1] = *reinterpret_cast<const v8h*>(p + 16);
    return u.v;
}

// ---------------- main dense GEMM:  C(Mx512) = A(Mx512)f16 @ W(512x512) + bias
// W is supplied TRANSPOSED (N-major: WT[n*512+k]) so every load is 16B contiguous.
// block = 256 threads = 8 waves as 4(M) x 2(N); block tile 128x128; wave tile 32x64.
__global__ __launch_bounds__(256)
void k_gemm_wmma(const _Float16* __restrict__ A, const _Float16* __restrict__ WT,
                 const float* __restrict__ bias, float* __restrict__ C) {
    __shared__ __align__(16) _Float16 sA [128 * LDSA];
    __shared__ __align__(16) _Float16 sBT[128 * LDSB];
    const int tid  = threadIdx.x;
    const int wave = tid >> 5;
    const int wm   = wave >> 1;          // 0..3
    const int wn   = wave & 1;           // 0..1
    const int bm0  = blockIdx.y * 128;
    const int bn0  = blockIdx.x * 128;
    v8f c[2][4] = {};

    for (int kk = 0; kk < T_DIM; kk += 32) {
        #pragma unroll
        for (int i = 0; i < 2; ++i) { // A tile 128x32 : 512 chunks of 8 halves
            int ci = tid + 256 * i;
            int r  = ci >> 2;
            int ch = (ci & 3) * 8;
            v8h v = *reinterpret_cast<const v8h*>(A + (size_t)(bm0 + r) * T_DIM + kk + ch);
            *reinterpret_cast<v8h*>(sA + r * LDSA + ch) = v;
        }
        #pragma unroll
        for (int i = 0; i < 2; ++i) { // B^T tile 128(cols)x32(k) : all b128
            int ci = tid + 256 * i;
            int cc = ci >> 2;
            int ch = (ci & 3) * 8;
            v8h v = *reinterpret_cast<const v8h*>(WT + (size_t)(bn0 + cc) * T_DIM + kk + ch);
            *reinterpret_cast<v8h*>(sBT + cc * LDSB + ch) = v;
        }
        __syncthreads();
        v16h a0 = load_a_frag(sA, wm * 32 + 0);
        v16h a1 = load_a_frag(sA, wm * 32 + 16);
        v16h b0 = load_b_frag(sBT, wn * 64 + 0);
        v16h b1 = load_b_frag(sBT, wn * 64 + 16);
        v16h b2 = load_b_frag(sBT, wn * 64 + 32);
        v16h b3 = load_b_frag(sBT, wn * 64 + 48);
        c[0][0] = __builtin_amdgcn_wmma_f32_16x16x32_f16(false, a0, false, b0, (short)0, c[0][0], false, false);
        c[0][1] = __builtin_amdgcn_wmma_f32_16x16x32_f16(false, a0, false, b1, (short)0, c[0][1], false, false);
        c[0][2] = __builtin_amdgcn_wmma_f32_16x16x32_f16(false, a0, false, b2, (short)0, c[0][2], false, false);
        c[0][3] = __builtin_amdgcn_wmma_f32_16x16x32_f16(false, a0, false, b3, (short)0, c[0][3], false, false);
        c[1][0] = __builtin_amdgcn_wmma_f32_16x16x32_f16(false, a1, false, b0, (short)0, c[1][0], false, false);
        c[1][1] = __builtin_amdgcn_wmma_f32_16x16x32_f16(false, a1, false, b1, (short)0, c[1][1], false, false);
        c[1][2] = __builtin_amdgcn_wmma_f32_16x16x32_f16(false, a1, false, b2, (short)0, c[1][2], false, false);
        c[1][3] = __builtin_amdgcn_wmma_f32_16x16x32_f16(false, a1, false, b3, (short)0, c[1][3], false, false);
        __syncthreads();
    }
    const int lane = tid & 31;
    const int col  = lane & 15;
    const int rb   = (lane >> 4) * 8;
    #pragma unroll
    for (int mi = 0; mi < 2; ++mi)
        #pragma unroll
        for (int ni = 0; ni < 4; ++ni) {
            int m0 = bm0 + wm * 32 + mi * 16;
            int n0 = bn0 + wn * 64 + ni * 16;
            v8f acc = c[mi][ni];
            #pragma unroll
            for (int r = 0; r < 8; ++r)
                C[(size_t)(m0 + rb + r) * T_DIM + n0 + col] = acc[r] + bias[n0 + col];
        }
}

// ---------------- batched adjacency GEMM:  V = mask@h + eps*h  (per sample)
// A = MP[b] (224x224 f16 padded), B^T = HPT[b] (512x224 f16, N-major), out f16 rows<200
// block 256 thr = 8 waves as 2(M) x 4(N); block tile 32x128; wave tile 16x32.
__global__ __launch_bounds__(256)
void k_agg_wmma(const _Float16* __restrict__ MPall, const _Float16* __restrict__ HPTall,
                const float* __restrict__ H, const float* __restrict__ epsp,
                _Float16* __restrict__ V16) {
    __shared__ __align__(16) _Float16 sA [32  * LDSA];
    __shared__ __align__(16) _Float16 sBT[128 * LDSB];
    const int b   = blockIdx.z;
    const _Float16* A  = MPall  + (size_t)b * ROI_PAD * ROI_PAD;
    const _Float16* BT = HPTall + (size_t)b * T_DIM * ROI_PAD;
    const int tid  = threadIdx.x;
    const int wave = tid >> 5;
    const int wm   = wave & 1;
    const int wn   = wave >> 1;
    const int bm0  = blockIdx.y * 32;
    const int bn0  = blockIdx.x * 128;
    v8f c[2] = {};

    for (int kk = 0; kk < ROI_PAD; kk += 32) {
        if (tid < 128) { // A tile 32x32 : 128 chunks
            int r  = tid >> 2;
            int ch = (tid & 3) * 8;
            v8h v = *reinterpret_cast<const v8h*>(A + (size_t)(bm0 + r) * ROI_PAD + kk + ch);
            *reinterpret_cast<v8h*>(sA + r * LDSA + ch) = v;
        }
        #pragma unroll
        for (int i = 0; i < 2; ++i) { // B^T tile 128(cols)x32(k) : all b128
            int ci = tid + 256 * i;
            int cc = ci >> 2;
            int ch = (ci & 3) * 8;
            v8h v = *reinterpret_cast<const v8h*>(BT + (size_t)(bn0 + cc) * ROI_PAD + kk + ch);
            *reinterpret_cast<v8h*>(sBT + cc * LDSB + ch) = v;
        }
        __syncthreads();
        v16h a0 = load_a_frag(sA, wm * 16);
        v16h b0 = load_b_frag(sBT, wn * 32 + 0);
        v16h b1 = load_b_frag(sBT, wn * 32 + 16);
        c[0] = __builtin_amdgcn_wmma_f32_16x16x32_f16(false, a0, false, b0, (short)0, c[0], false, false);
        c[1] = __builtin_amdgcn_wmma_f32_16x16x32_f16(false, a0, false, b1, (short)0, c[1], false, false);
        __syncthreads();
    }
    const float ev  = *epsp;
    const int lane = tid & 31;
    const int col  = lane & 15;
    const int rb   = (lane >> 4) * 8;
    #pragma unroll
    for (int ni = 0; ni < 2; ++ni) {
        int n0 = bn0 + wn * 32 + ni * 16;
        v8f acc = c[ni];
        #pragma unroll
        for (int r = 0; r < 8; ++r) {
            int mg = bm0 + wm * 16 + rb + r;   // row within sample (padded to 224)
            if (mg < ROI) {
                size_t idx = ((size_t)b * ROI + mg) * T_DIM + n0 + col;
                V16[idx] = (_Float16)(acc[r] + ev * H[idx]);
            }
        }
    }
}

// ---------------- staging kernels ----------------
__global__ __launch_bounds__(256)
void k_maskpad(const float* __restrict__ a, _Float16* __restrict__ MP) {
    size_t i = (size_t)blockIdx.x * 256 + threadIdx.x;   // over 256*224*224
    if (i >= (size_t)BATCH * ROI_PAD * ROI_PAD) return;
    int cc = (int)(i % ROI_PAD);
    int rr = (int)((i / ROI_PAD) % ROI_PAD);
    int bb = (int)(i / (ROI_PAD * ROI_PAD));
    _Float16 v = (_Float16)0.f;
    if (rr < ROI && cc < ROI)
        v = (a[((size_t)bb * ROI + rr) * ROI + cc] != 0.f) ? (_Float16)1.f : (_Float16)0.f;
    MP[i] = v;
}

// h (b,200,512) f32 -> HPT (b,512,224) f16, transposed + K-padded, via LDS tile.
// grid (ROI_PAD/32, BATCH); each block: 32 k-rows x 512 n-cols of one sample.
__global__ __launch_bounds__(256)
void k_htrans(const float* __restrict__ H, _Float16* __restrict__ HPT) {
    __shared__ __align__(16) _Float16 sT[32 * LDST];     // 33280 B
    const int b  = blockIdx.y;
    const int k0 = blockIdx.x * 32;
    for (int i = threadIdx.x; i < 32 * 128; i += 256) {  // vec4 units, coalesced read
        int r  = i >> 7;           // 0..31
        int c4 = i & 127;          // 0..127
        int k  = k0 + r;
        v4h o = {};
        if (k < ROI) {
            float4 x = reinterpret_cast<const float4*>(H)[((size_t)b * ROI + k) * 128 + c4];
            o[0] = (_Float16)x.x; o[1] = (_Float16)x.y;
            o[2] = (_Float16)x.z; o[3] = (_Float16)x.w;
        }
        *reinterpret_cast<v4h*>(sT + r * LDST + c4 * 4) = o;
    }
    __syncthreads();
    for (int ci = threadIdx.x; ci < 512 * 4; ci += 256) { // b128 writes, K-contiguous
        int n  = ci >> 2;
        int ch = (ci & 3) * 8;
        v8h o;
        #pragma unroll
        for (int j = 0; j < 8; ++j) o[j] = sT[(ch + j) * LDST + n];
        *reinterpret_cast<v8h*>(HPT + ((size_t)b * T_DIM + n) * ROI_PAD + k0 + ch) = o;
    }
}

// W (512x512 f32, K-major) -> WT16 (512x512 f16, N-major)
__global__ __launch_bounds__(256)
void k_wtrans(const float* __restrict__ W, _Float16* __restrict__ WT) {
    int i = blockIdx.x * 256 + threadIdx.x;   // i = n*512 + k
    int n = i >> 9, k = i & 511;
    WT[i] = (_Float16)W[k * T_DIM + n];
}

// ---------------- BN statistics (deterministic two-stage) ----------------
__global__ __launch_bounds__(256)
void k_stats_part(const float* __restrict__ X, float* __restrict__ ps, float* __restrict__ pq) {
    int c  = threadIdx.x;                 // handles cols c and c+256
    int r0 = blockIdx.x * 128;
    float s0 = 0.f, s1 = 0.f, q0 = 0.f, q1 = 0.f;
    for (int r = 0; r < 128; ++r) {
        float a = X[(size_t)(r0 + r) * T_DIM + c];
        float b = X[(size_t)(r0 + r) * T_DIM + c + 256];
        s0 += a; q0 += a * a; s1 += b; q1 += b * b;
    }
    ps[blockIdx.x * T_DIM + c]       = s0;
    ps[blockIdx.x * T_DIM + c + 256] = s1;
    pq[blockIdx.x * T_DIM + c]       = q0;
    pq[blockIdx.x * T_DIM + c + 256] = q1;
}

__global__ __launch_bounds__(256)
void k_stats_fin(const float* __restrict__ ps, const float* __restrict__ pq,
                 const float* __restrict__ g, const float* __restrict__ bb,
                 float* __restrict__ ss) {
    int c = blockIdx.x * 256 + threadIdx.x;              // 0..511
    float s = 0.f, q = 0.f;
    for (int i = 0; i < 400; ++i) { s += ps[i * T_DIM + c]; q += pq[i * T_DIM + c]; }
    const float invN = 1.f / (float)MROWS;
    float mean = s * invN;
    float var  = q * invN - mean * mean;
    float sc   = g[c] * rsqrtf(var + BN_EPS);
    ss[c]         = sc;
    ss[T_DIM + c] = bb[c] - mean * sc;
}

__device__ __forceinline__ float elu1(float y) { return y > 0.f ? y : (expf(y) - 1.f); }

__global__ __launch_bounds__(256)
void k_bn_elu_h(const float4* __restrict__ X, const float* __restrict__ ss,
                v4h* __restrict__ Y) {
    int i = blockIdx.x * 256 + threadIdx.x;              // 51200*512/4 vec4
    int c = (i * 4) & (T_DIM - 1);
    float4 x = X[i];
    float y0 = elu1(x.x * ss[c]     + ss[T_DIM + c]);
    float y1 = elu1(x.y * ss[c + 1] + ss[T_DIM + c + 1]);
    float y2 = elu1(x.z * ss[c + 2] + ss[T_DIM + c + 2]);
    float y3 = elu1(x.w * ss[c + 3] + ss[T_DIM + c + 3]);
    v4h o; o[0] = (_Float16)y0; o[1] = (_Float16)y1; o[2] = (_Float16)y2; o[3] = (_Float16)y3;
    Y[i] = o;
}

__global__ __launch_bounds__(256)
void k_bn_elu_f(const float4* __restrict__ X, const float* __restrict__ ss,
                float4* __restrict__ Y) {
    int i = blockIdx.x * 256 + threadIdx.x;
    int c = (i * 4) & (T_DIM - 1);
    float4 x = X[i];
    float4 o;
    o.x = elu1(x.x * ss[c]     + ss[T_DIM + c]);
    o.y = elu1(x.y * ss[c + 1] + ss[T_DIM + c + 1]);
    o.z = elu1(x.z * ss[c + 2] + ss[T_DIM + c + 2]);
    o.w = elu1(x.w * ss[c + 3] + ss[T_DIM + c + 3]);
    Y[i] = o;
}

// ---------------- readout ----------------
__global__ __launch_bounds__(512)
void k_roimean(const float* __restrict__ H, float* __restrict__ XM) {
    int b = blockIdx.x, t = threadIdx.x;                 // 512 threads
    float s = 0.f;
    for (int r = 0; r < ROI; ++r) s += H[((size_t)b * ROI + r) * T_DIM + t];
    XM[b * T_DIM + t] = s * (1.f / (float)ROI);
}

__global__ __launch_bounds__(256)
void k_mlp1(const float* __restrict__ XM, const float* __restrict__ W,
            const float* __restrict__ bm, float* __restrict__ Y1) {
    int row = blockIdx.x, col = threadIdx.x;             // 256x256
    float s = 0.f;
    for (int k = 0; k < T_DIM; ++k) s += XM[row * T_DIM + k] * W[k * 256 + col];
    Y1[row * 256 + col] = s + bm[col];
}

__global__ __launch_bounds__(256)
void k_bn_relu(const float* __restrict__ Y1, const float* __restrict__ g,
               const float* __restrict__ bb, float* __restrict__ Z) {
    int c = blockIdx.x, r = threadIdx.x;                 // col, row (256 each)
    float v = Y1[r * 256 + c];
    __shared__ float s1[256], s2[256];
    s1[r] = v; s2[r] = v * v;
    __syncthreads();
    for (int off = 128; off > 0; off >>= 1) {
        if (r < off) { s1[r] += s1[r + off]; s2[r] += s2[r + off]; }
        __syncthreads();
    }
    float mean = s1[0] * (1.f / 256.f);
    float var  = s2[0] * (1.f / 256.f) - mean * mean;
    float sc   = g[c] * rsqrtf(var + BN_EPS);
    float y    = v * sc + (bb[c] - mean * sc);
    Z[r * 256 + c] = y > 0.f ? y : 0.f;
}

__global__ __launch_bounds__(256)
void k_mlp2(const float* __restrict__ Z, const float* __restrict__ W,
            const float* __restrict__ bm, float* __restrict__ out) {
    int tid = blockIdx.x * 256 + threadIdx.x;            // 0..511
    int b = tid >> 1, j = tid & 1;
    float s = 0.f;
    for (int k = 0; k < 256; ++k) s += Z[b * 256 + k] * W[k * 2 + j];
    out[b * 2 + j] = s + bm[j];
}

// ---------------- orchestration ----------------
extern "C" void kernel_launch(void* const* d_in, const int* in_sizes, int n_in,
                              void* d_out, int out_size, void* d_ws, size_t ws_size,
                              hipStream_t stream) {
    (void)in_sizes; (void)n_in; (void)out_size; (void)ws_size;
    const float* x    = (const float*)d_in[0];
    const float* a    = (const float*)d_in[1];
    const float* eps  = (const float*)d_in[2];
    const float* W1   = (const float*)d_in[3];
    const float* b1   = (const float*)d_in[4];
    const float* g1   = (const float*)d_in[5];
    const float* be1  = (const float*)d_in[6];
    const float* W2   = (const float*)d_in[7];
    const float* b2   = (const float*)d_in[8];
    const float* g2   = (const float*)d_in[9];
    const float* be2  = (const float*)d_in[10];
    const float* gl   = (const float*)d_in[11];
    const float* bl   = (const float*)d_in[12];
    const float* Wm1  = (const float*)d_in[13];
    const float* bm1  = (const float*)d_in[14];
    const float* gm   = (const float*)d_in[15];
    const float* betam= (const float*)d_in[16];
    const float* Wm2  = (const float*)d_in[17];
    const float* bm2  = (const float*)d_in[18];

    char* ws = (char*)d_ws;
    float*    H    = (float*)(ws + OFF_H);
    float*    U    = (float*)(ws + OFF_U);
    _Float16* HPT  = (_Float16*)(ws + OFF_U);    // aliases U (disjoint in time)
    _Float16* V16  = (_Float16*)(ws + OFF_V16);  // also serves as T16
    _Float16* MP   = (_Float16*)(ws + OFF_MP);
    _Float16* WT16 = (_Float16*)(ws + OFF_W16);
    float*    PS   = (float*)(ws + OFF_PS);
    float*    PQ   = (float*)(ws + OFF_PQ);
    float*    SS   = (float*)(ws + OFF_SS);
    float*    XM   = (float*)(ws + OFF_XM);
    float*    Y1   = (float*)(ws + OFF_Y1);
    float*    Z    = (float*)(ws + OFF_Z);

    const int NVEC   = MROWS * T_DIM / 4;        // 6,553,600
    const int G_BN   = NVEC / 256;               // 25600
    const int G_MASK = (BATCH * ROI_PAD * ROI_PAD + 255) / 256;

    hipMemcpyAsync(H, x, (size_t)MROWS * T_DIM * sizeof(float),
                   hipMemcpyDeviceToDevice, stream);
    k_maskpad<<<G_MASK, 256, 0, stream>>>(a, MP);

    for (int l = 0; l < NLAYERS; ++l) {
        // agg = mask@h ; v = agg + eps*h   (f16 out)
        k_htrans<<<dim3(ROI_PAD / 32, BATCH), 256, 0, stream>>>(H, HPT);
        k_agg_wmma<<<dim3(T_DIM / 128, ROI_PAD / 32, BATCH), 256, 0, stream>>>(
            MP, HPT, H, eps + l, V16);
        // u = v @ W1 + b1 ; BN ; elu  (f16 out -> T16 alias V16)
        k_wtrans<<<1024, 256, 0, stream>>>(W1 + (size_t)l * T_DIM * T_DIM, WT16);
        k_gemm_wmma<<<dim3(T_DIM / 128, MROWS / 128), 256, 0, stream>>>(
            V16, WT16, b1 + l * T_DIM, U);
        k_stats_part<<<400, 256, 0, stream>>>(U, PS, PQ);
        k_stats_fin<<<2, 256, 0, stream>>>(PS, PQ, g1 + l * T_DIM, be1 + l * T_DIM, SS);
        k_bn_elu_h<<<G_BN, 256, 0, stream>>>((const float4*)U, SS, (v4h*)V16);
        // u = t @ W2 + b2 ; BN ; elu  (f32 out -> H)
        k_wtrans<<<1024, 256, 0, stream>>>(W2 + (size_t)l * T_DIM * T_DIM, WT16);
        k_gemm_wmma<<<dim3(T_DIM / 128, MROWS / 128), 256, 0, stream>>>(
            V16, WT16, b2 + l * T_DIM, U);
        k_stats_part<<<400, 256, 0, stream>>>(U, PS, PQ);
        k_stats_fin<<<2, 256, 0, stream>>>(PS, PQ, g2 + l * T_DIM, be2 + l * T_DIM, SS);
        k_bn_elu_f<<<G_BN, 256, 0, stream>>>((const float4*)U, SS, (float4*)H);
        // h = elu(BN(v))  (in-place on H)
        k_stats_part<<<400, 256, 0, stream>>>(H, PS, PQ);
        k_stats_fin<<<2, 256, 0, stream>>>(PS, PQ, gl + l * T_DIM, bl + l * T_DIM, SS);
        k_bn_elu_f<<<G_BN, 256, 0, stream>>>((const float4*)H, SS, (float4*)H);
    }

    // readout MLP
    k_roimean<<<BATCH, 512, 0, stream>>>(H, XM);
    k_mlp1<<<BATCH, 256, 0, stream>>>(XM, Wm1, bm1, Y1);
    k_bn_relu<<<256, 256, 0, stream>>>(Y1, gm, betam, Z);
    k_mlp2<<<2, 256, 0, stream>>>(Z, Wm2, bm2, (float*)d_out);
}
